// recovery_net_88192858456629
// MI455X (gfx1250) — matrix operand order repaired and checked by reading, not compiled, not run on hardware
//
#include <hip/hip_runtime.h>
#include <hip/hip_bf16.h>

typedef __attribute__((ext_vector_type(16))) _Float16 v16h;
typedef __attribute__((ext_vector_type(8)))  float    v8f;

#define B_ROWS    131072
#define TILE_ROWS 128
#define NCAT      29
#define BN_EPS    1e-5f

// ---- WMMA fragment helpers (contiguous k order: lane half h holds k = 16h..16h+15, A and B alike) ----

// A fragment 16x32 f16 from row-major LDS tile X[row][ldx].
// lanes 0-15: M=lane, K = k0+0..15; lanes 16-31: M=lane-16, K = k0+16..31
__device__ __forceinline__ v16h load_a_lds(const _Float16* X, int ldx, int row0, int k0) {
    const int lane = threadIdx.x & 31;
    const int m    = lane & 15;
    const int koff = (lane >> 4) << 4;   // 0 or 16
    const _Float16* p = X + (row0 + m) * ldx + k0 + koff;
    v16h a;
#pragma unroll
    for (int j = 0; j < 8; ++j) a[j]     = p[j];
#pragma unroll
    for (int j = 0; j < 8; ++j) a[8 + j] = p[8 + j];
    return a;
}

// Direct (fallback) B fragment 32x16 from row-major f32 global weights:
// 16 scattered dword gathers + cvt. Used only if d_ws is too small.
__device__ __forceinline__ v16h load_b_gl(const float* __restrict__ W, int ldw,
                                          int k0, int col0, int kmax) {
    const int lane = threadIdx.x & 31;
    const int n    = lane & 15;
    const int kb   = k0 + ((lane >> 4) << 4);
    v16h b;
#pragma unroll
    for (int j = 0; j < 16; ++j) {
        int   k = kb + j;
        float v = (k < kmax) ? W[k * ldw + col0 + n] : 0.0f;
        b[j] = (_Float16)v;
    }
    return b;
}

// Packed B fragment: 512 f16 per fragment, [frag][lane][16]; each lane reads its
// 32 contiguous bytes -> two global_load_b128, no conversion VALU in the hot loop.
__device__ __forceinline__ v16h load_b_pk(const _Float16* __restrict__ base, int frag) {
    const int lane = threadIdx.x & 31;
    return *(const v16h*)(base + (size_t)frag * 512 + lane * 16);
}

__device__ __forceinline__ v8f wmma_f16(v16h a, v16h b, v8f c) {
    // 8 args: (neg_a, A, neg_b, B, c_mod, C, reuse_a, reuse_b)
    return __builtin_amdgcn_wmma_f32_16x16x32_f16(false, a, false, b, (short)0, c, false, false);
}

// ---- weight packing kernel: f32 row-major -> f16 WMMA-B fragment layout ----
// One 32-lane block per fragment. frag = ntile * nk + ks; K slice = ks*32..+31.
__global__ __launch_bounds__(32) void pack_b_frags(const float* __restrict__ W, int ldw,
                                                   int kmax, int nk,
                                                   _Float16* __restrict__ dst) {
    const int frag = blockIdx.x;
    const int nt   = frag / nk;
    const int ks   = frag % nk;
    const int lane = threadIdx.x & 31;
    const int n    = lane & 15;
    const int kb   = ks * 32 + ((lane >> 4) << 4);
    const int col  = nt * 16 + n;
    _Float16* p = dst + (size_t)frag * 512 + lane * 16;
#pragma unroll
    for (int j = 0; j < 16; ++j) {
        int   k = kb + j;
        float v = (k < kmax) ? W[k * ldw + col] : 0.0f;
        p[j] = (_Float16)v;
    }
}

template <bool PK>
__global__ __launch_bounds__(256) void fused_recovery_net(
    const float* __restrict__ x,
    const float* __restrict__ E,   const float* __restrict__ W1,
    const float* __restrict__ b1,  const float* __restrict__ W2,
    const float* __restrict__ b2,
    const float* __restrict__ rW1, const float* __restrict__ rb1,
    const float* __restrict__ g1,  const float* __restrict__ be1,
    const float* __restrict__ m1,  const float* __restrict__ v1,
    const float* __restrict__ rW2, const float* __restrict__ rb2,
    const float* __restrict__ g2,  const float* __restrict__ be2,
    const float* __restrict__ m2,  const float* __restrict__ v2,
    const float* __restrict__ rW3, const float* __restrict__ rb3,
    const float* __restrict__ rW4, const float* __restrict__ rb4,
    const _Float16* __restrict__ pW1,   // packed rW1 frags: 48 ntiles x 2 ksteps
    const _Float16* __restrict__ pW2,   // packed rW2 frags: 4 ntiles x 24 ksteps
    const _Float16* __restrict__ pW3,   // packed rW3 frags: 1 ntile x 2 ksteps
    float* __restrict__ out)
{
    // 48 KB static LDS total
    __shared__ _Float16 smA[TILE_ROWS * 64];    // xin (f16, K padded 36->64); later reused for h2
    __shared__ _Float16 smH[TILE_ROWS * 128];   // h1 column chunk (128 cols of 768)

    const int t    = threadIdx.x;
    const int lane = t & 31;
    const int wv   = t >> 5;                    // 8 waves
    const int row0 = blockIdx.x * TILE_ROWS;

    // ---------------- Stage 0: categorical MLPs + xin assembly (VALU) --------------
    {
        const int r    = t >> 1;                // 0..127, 2 threads per row
        const int half = t & 1;
        const float* xr  = x + (size_t)(row0 + r) * 36;
        _Float16*    xin = smA + r * 64;
        if (half == 0) {
#pragma unroll
            for (int j = 0; j < 7; ++j) xin[j] = (_Float16)xr[j];
            for (int j = 36; j < 50; ++j) xin[j] = (_Float16)0.0f;   // K padding
        } else {
            for (int j = 50; j < 64; ++j) xin[j] = (_Float16)0.0f;   // K padding
        }
        const int c0 = half ? 15 : 0;
        const int c1 = half ? NCAT : 15;
        for (int c = c0; c < c1; ++c) {
            const int idx = (int)xr[7 + c];
            const float* e  = E + ((c << 2) + idx) * 12;   // E[c][idx][:]
            float emb[12];
#pragma unroll
            for (int d = 0; d < 12; ++d) emb[d] = e[d];
            const float* w1 = W1 + c * 144;                // W1[c][d][e]
            const float* bb = b1 + c * 12;
            const float* w2 = W2 + c * 12;
            float o = b2[c];
#pragma unroll
            for (int ee = 0; ee < 12; ++ee) {
                float h = bb[ee];
#pragma unroll
                for (int d = 0; d < 12; ++d) h = fmaf(emb[d], w1[d * 12 + ee], h);
                h = fmaxf(h, 0.0f);
                o = fmaf(h, w2[ee], o);
            }
            xin[7 + c] = (_Float16)fmaxf(o, 0.0f);
        }
    }
    __syncthreads();

    // ---------------- Layers 1+2 fused over 6 chunks of 128 h1-columns -------------
    // layer2 ownership: wave -> (N-tile n2 of 64 cols, 64-row group mg)
    const int n2 = wv & 3;
    const int mg = wv >> 2;
    v8f acc[4] = {};                            // persistent layer-2 accumulators

    for (int ci = 0; ci < 6; ++ci) {
        // ---- layer1: this wave computes N-tile (ci*8 + wv) of h1 ----
        {
            const int nt   = ci * 8 + wv;
            const int col0 = nt * 16;
            v16h bf0, bf1;
            if (PK) {
                bf0 = load_b_pk(pW1, nt * 2 + 0);
                bf1 = load_b_pk(pW1, nt * 2 + 1);
            } else {
                bf0 = load_b_gl(rW1, 768, 0,  col0, 36);
                bf1 = load_b_gl(rW1, 768, 32, col0, 36);
            }
            const int   col   = col0 + (lane & 15);
            const float scale = g1[col] * rsqrtf(v1[col] + BN_EPS);
            const float affb  = (rb1[col] - m1[col]) * scale + be1[col];
            const int   lcol  = col - ci * 128;            // column within smH chunk
#pragma unroll
            for (int mt = 0; mt < 8; ++mt) {
                v16h a0 = load_a_lds(smA, 64, mt * 16, 0);
                v16h a1 = load_a_lds(smA, 64, mt * 16, 32);
                v8f  c  = {};
                c = wmma_f16(a0, bf0, c);
                c = wmma_f16(a1, bf1, c);
#pragma unroll
                for (int r = 0; r < 8; ++r) {
                    float h = fmaf(c[r], scale, affb);     // bias + batchnorm affine
                    h = fmaxf(h, 0.0f);                    // relu
                    int rr = mt * 16 + r + ((lane >> 4) << 3);
                    smH[rr * 128 + lcol] = (_Float16)h;
                }
            }
        }
        __syncthreads();

        // ---- layer2 partial: accumulate this 128-wide K slice ----
        {
#pragma unroll
            for (int ks = 0; ks < 4; ++ks) {
                v16h bf;
                if (PK) {
                    bf = load_b_pk(pW2, n2 * 24 + (ci * 4 + ks));
                } else {
                    bf = load_b_gl(rW2, 64, ci * 128 + ks * 32, n2 * 16, 768);
                }
#pragma unroll
                for (int mt = 0; mt < 4; ++mt) {
                    v16h af = load_a_lds(smH, 128, mg * 64 + mt * 16, ks * 32);
                    acc[mt] = wmma_f16(af, bf, acc[mt]);
                }
            }
        }
        __syncthreads();
    }

    // ---------------- Layer 2 epilogue: BN + ReLU -> h2 (f16, reuse smA) -----------
    {
        const int   col   = n2 * 16 + (lane & 15);
        const float scale = g2[col] * rsqrtf(v2[col] + BN_EPS);
        const float affb  = (rb2[col] - m2[col]) * scale + be2[col];
#pragma unroll
        for (int mt = 0; mt < 4; ++mt) {
#pragma unroll
            for (int r = 0; r < 8; ++r) {
                float h = fmaf(acc[mt][r], scale, affb);
                h = fmaxf(h, 0.0f);
                int rr = mg * 64 + mt * 16 + r + ((lane >> 4) << 3);
                smA[rr * 64 + col] = (_Float16)h;
            }
        }
    }
    __syncthreads();

    // ---------------- Layers 3+4: wave wv owns rows wv*16..wv*16+15 ----------------
    {
        v16h a0 = load_a_lds(smA, 64, wv * 16, 0);
        v16h a1 = load_a_lds(smA, 64, wv * 16, 32);
        v16h bf0, bf1;
        if (PK) {
            bf0 = load_b_pk(pW3, 0);
            bf1 = load_b_pk(pW3, 1);
        } else {
            bf0 = load_b_gl(rW3, 16, 0,  0, 64);
            bf1 = load_b_gl(rW3, 16, 32, 0, 64);
        }
        v8f c = {};
        c = wmma_f16(a0, bf0, c);
        c = wmma_f16(a1, bf1, c);

        const int   n  = lane & 15;
        const float w4 = rW4[n];
        const float b3 = rb3[n];
        const float b4 = rb4[0];
#pragma unroll
        for (int r = 0; r < 8; ++r) {
            float h3 = fmaxf(c[r] + b3, 0.0f);             // layer3 relu
            float v  = h3 * w4;                            // layer4 partial product
            v += __shfl_xor(v, 1, 32);                     // 16-lane butterfly sum
            v += __shfl_xor(v, 2, 32);
            v += __shfl_xor(v, 4, 32);
            v += __shfl_xor(v, 8, 32);
            if ((lane & 15) == 0) {
                float z = v + b4;
                float o = 1.0f / (1.0f + __expf(-z));      // sigmoid
                int rr = row0 + wv * 16 + r + ((lane >> 4) << 3);
                out[rr] = o;
            }
        }
    }
}

extern "C" void kernel_launch(void* const* d_in, const int* in_sizes, int n_in,
                              void* d_out, int out_size, void* d_ws, size_t ws_size,
                              hipStream_t stream) {
    (void)in_sizes; (void)n_in; (void)out_size;
    const float* x   = (const float*)d_in[0];
    const float* E   = (const float*)d_in[1];
    const float* W1  = (const float*)d_in[2];
    const float* b1  = (const float*)d_in[3];
    const float* W2  = (const float*)d_in[4];
    const float* b2  = (const float*)d_in[5];
    const float* rW1 = (const float*)d_in[6];
    const float* rb1 = (const float*)d_in[7];
    const float* g1  = (const float*)d_in[8];
    const float* be1 = (const float*)d_in[9];
    const float* m1  = (const float*)d_in[10];
    const float* v1  = (const float*)d_in[11];
    const float* rW2 = (const float*)d_in[12];
    const float* rb2 = (const float*)d_in[13];
    const float* g2  = (const float*)d_in[14];
    const float* be2 = (const float*)d_in[15];
    const float* m2  = (const float*)d_in[16];
    const float* v2  = (const float*)d_in[17];
    const float* rW3 = (const float*)d_in[18];
    const float* rb3 = (const float*)d_in[19];
    const float* rW4 = (const float*)d_in[20];
    const float* rb4 = (const float*)d_in[21];
    float* out = (float*)d_out;

    // Packed B-fragment workspace: rW1 48x2 frags, rW2 4x24 frags, rW3 1x2 frags,
    // 512 f16 (1 KB) each -> 194 KB total.
    const int    NFRAG1 = 48 * 2, NFRAG2 = 4 * 24, NFRAG3 = 2;
    const size_t need   = (size_t)(NFRAG1 + NFRAG2 + NFRAG3) * 512 * sizeof(_Float16);
    const dim3   grid(B_ROWS / TILE_ROWS);   // 1024 blocks
    const dim3   block(256);                 // 8 wave32 waves

    if (ws_size >= need) {
        _Float16* pW1 = (_Float16*)d_ws;
        _Float16* pW2 = pW1 + (size_t)NFRAG1 * 512;
        _Float16* pW3 = pW2 + (size_t)NFRAG2 * 512;
        pack_b_frags<<<NFRAG1, 32, 0, stream>>>(rW1, 768, 36,  2,  pW1);
        pack_b_frags<<<NFRAG2, 32, 0, stream>>>(rW2, 64,  768, 24, pW2);
        pack_b_frags<<<NFRAG3, 32, 0, stream>>>(rW3, 16,  64,  2,  pW3);
        fused_recovery_net<true><<<grid, block, 0, stream>>>(
            x, E, W1, b1, W2, b2, rW1, rb1, g1, be1, m1, v1,
            rW2, rb2, g2, be2, m2, v2, rW3, rb3, rW4, rb4,
            pW1, pW2, pW3, out);
    } else {
        fused_recovery_net<false><<<grid, block, 0, stream>>>(
            x, E, W1, b1, W2, b2, rW1, rb1, g1, be1, m1, v1,
            rW2, rb2, g2, be2, m2, v2, rW3, rb3, rW4, rb4,
            nullptr, nullptr, nullptr, out);
    }
}
